// i3HybridModel_61272003444765
// MI455X (gfx1250) — matrix operand order repaired and checked by reading, not compiled, not run on hardware
//
#include <hip/hip_runtime.h>
#include <math.h>

// ---------------- model dims ----------------
#define DMODEL 768
#define NHEAD  12
#define HEADD  64
#define VOCAB  32000
#define BBATCH 2
#define TSEQ   1024
#define FFDIM  3072

// ---------------- types ----------------
typedef __bf16 bf16_t;
typedef bf16_t v16bf __attribute__((ext_vector_type(16)));
typedef float  v8f   __attribute__((ext_vector_type(8)));

// ---------------- CDNA5 async global->LDS (ASYNCcnt path) ----------------
__device__ __forceinline__ void async_b128(unsigned lds_off, const void* gaddr) {
    asm volatile("global_load_async_to_lds_b128 %0, %1, off"
                 :: "v"(lds_off), "v"((unsigned long long)(size_t)gaddr)
                 : "memory");
}
__device__ __forceinline__ void async_wait0() {
    asm volatile("s_wait_asynccnt 0" ::: "memory");
}

// ---------------- block reductions (wave32) ----------------
__device__ __forceinline__ float warpSum(float v) {
    #pragma unroll
    for (int o = 16; o > 0; o >>= 1) v += __shfl_down(v, o, 32);
    return v;
}
__device__ __forceinline__ float warpMax(float v) {
    #pragma unroll
    for (int o = 16; o > 0; o >>= 1) v = fmaxf(v, __shfl_down(v, o, 32));
    return v;
}
__device__ __forceinline__ float blockSum(float v) {
    __shared__ float sh[9];
    int lane = threadIdx.x & 31, w = threadIdx.x >> 5;
    v = warpSum(v);
    if (lane == 0) sh[w] = v;
    __syncthreads();
    if (w == 0) {
        float t = (lane < (int)((blockDim.x + 31) >> 5)) ? sh[lane] : 0.0f;
        t = warpSum(t);
        if (lane == 0) sh[8] = t;
    }
    __syncthreads();
    float r = sh[8];
    __syncthreads();
    return r;
}
__device__ __forceinline__ float blockMax(float v) {
    __shared__ float sh[9];
    int lane = threadIdx.x & 31, w = threadIdx.x >> 5;
    v = warpMax(v);
    if (lane == 0) sh[w] = v;
    __syncthreads();
    if (w == 0) {
        float t = (lane < (int)((blockDim.x + 31) >> 5)) ? sh[lane] : -3.0e38f;
        t = warpMax(t);
        if (lane == 0) sh[8] = t;
    }
    __syncthreads();
    float r = sh[8];
    __syncthreads();
    return r;
}

// ---------------- fp32 -> bf16 conversion ----------------
__global__ __launch_bounds__(256)
void f32_to_bf16_kernel(const float* __restrict__ src, bf16_t* __restrict__ dst,
                        long long n) {
    long long gid = (long long)blockIdx.x * blockDim.x + threadIdx.x;
    if (gid < n) dst[gid] = (bf16_t)src[gid];
}

// ---------------- embedding + positional ----------------
__global__ __launch_bounds__(256)
void embed_kernel(const int* __restrict__ idx, const float* __restrict__ emb,
                  const float* __restrict__ pos, float* __restrict__ x,
                  int T, int D, long long n) {
    long long gid = (long long)blockIdx.x * blockDim.x + threadIdx.x;
    if (gid >= n) return;
    int c  = (int)(gid % D);
    long long bt = gid / D;
    int t  = (int)(bt % T);
    x[gid] = emb[(long long)idx[bt] * D + c] + pos[(long long)t * D + c];
}

// ---------------- layernorm: fp32 in -> bf16 out ----------------
__global__ __launch_bounds__(256)
void ln_kernel(const float* __restrict__ x, const float* __restrict__ g,
               const float* __restrict__ b, bf16_t* __restrict__ out, int D) {
    long long row = blockIdx.x;
    const float* xr = x + row * D;
    float s = 0.0f;
    for (int c = threadIdx.x; c < D; c += blockDim.x) s += xr[c];
    float mean = blockSum(s) / (float)D;
    float vs = 0.0f;
    for (int c = threadIdx.x; c < D; c += blockDim.x) {
        float d = xr[c] - mean; vs += d * d;
    }
    float rstd = rsqrtf(blockSum(vs) / (float)D + 1e-5f);
    bf16_t* orow = out + row * D;
    for (int c = threadIdx.x; c < D; c += blockDim.x)
        orow[c] = (bf16_t)((xr[c] - mean) * rstd * g[c] + b[c]);
}

// ---------------- time/channel shift mix (bf16 in/out) ----------------
__global__ __launch_bounds__(256)
void mix_kernel(const bf16_t* __restrict__ h,
                const float* __restrict__ tmk, const float* __restrict__ tmv,
                const float* __restrict__ tmr,
                bf16_t* __restrict__ ok, bf16_t* __restrict__ ov,
                bf16_t* __restrict__ orr,
                int T, int D, long long n) {
    long long gid = (long long)blockIdx.x * blockDim.x + threadIdx.x;
    if (gid >= n) return;
    int c  = (int)(gid % D);
    long long bt = gid / D;
    int t  = (int)(bt % T);
    float hc = (float)h[gid];
    float hp = (t == 0) ? 0.0f : (float)h[gid - D];
    float a = tmk[c]; ok[gid] = (bf16_t)(hc * a + hp * (1.0f - a));
    float bc = tmv[c]; ov[gid] = (bf16_t)(hc * bc + hp * (1.0f - bc));
    if (tmr) { float r = tmr[c]; orr[gid] = (bf16_t)(hc * r + hp * (1.0f - r)); }
}

// ---------------- V transpose: qkv(B,T,3D) -> vt(B*H, HD, T) bf16 ----------------
__global__ __launch_bounds__(256)
void vtrans_kernel(const bf16_t* __restrict__ qkv, bf16_t* __restrict__ vt,
                   int T, int D, int H, int HD, long long n) {
    long long gid = (long long)blockIdx.x * blockDim.x + threadIdx.x;
    if (gid >= n) return;
    int t  = (int)(gid % T);
    long long r = gid / T;
    int nd = (int)(r % HD);
    long long bh = r / HD;
    long long b = bh / H;
    int h = (int)(bh % H);
    vt[gid] = qkv[(b * T + t) * 3LL * D + 2 * D + (long long)h * HD + nd];
}

// ---------------- WKV sequential recurrence (fp32) ----------------
__global__ __launch_bounds__(256)
void wkv_kernel(const float* __restrict__ k, const float* __restrict__ v,
                const float* __restrict__ td, const float* __restrict__ tf,
                float* __restrict__ y, int B, int T, int D) {
    int gid = blockIdx.x * blockDim.x + threadIdx.x;
    if (gid >= B * D) return;
    int b = gid / D, c = gid % D;
    float w = -expf(td[c]);
    float u = tf[c];
    float aa = 0.0f, bb = 0.0f, pp = -1.0e30f;
    const float* kp = k + (long long)b * T * D + c;
    const float* vp = v + (long long)b * T * D + c;
    float*       yp = y + (long long)b * T * D + c;
    for (int t = 0; t < T; ++t) {
        float kt = kp[(long long)t * D];
        float vt = vp[(long long)t * D];
        float ww = u + pp;
        float p  = fmaxf(ww, kt);
        float e1 = expf(ww - p), e2 = expf(kt - p);
        yp[(long long)t * D] = (aa * e1 + vt * e2) / (bb * e1 + e2 + 1e-8f);
        float ww2 = w + pp;
        float p2  = fmaxf(ww2, kt);
        float e1b = expf(ww2 - p2), e2b = expf(kt - p2);
        aa = aa * e1b + vt * e2b;
        bb = bb * e1b + e2b;
        pp = p2;
    }
}

// ---------------- elementwise epilogues ----------------
__global__ __launch_bounds__(256)
void sigmul_bf16_kernel(const float* __restrict__ r, const float* __restrict__ y,
                        bf16_t* __restrict__ out, long long n) {
    long long gid = (long long)blockIdx.x * blockDim.x + threadIdx.x;
    if (gid >= n) return;
    float s = 1.0f / (1.0f + expf(-r[gid]));
    out[gid] = (bf16_t)(s * y[gid]);
}
__global__ __launch_bounds__(256)
void sigmul_acc_kernel(const float* __restrict__ r, const float* __restrict__ y,
                       float* __restrict__ out, long long n) {
    long long gid = (long long)blockIdx.x * blockDim.x + threadIdx.x;
    if (gid >= n) return;
    float s = 1.0f / (1.0f + expf(-r[gid]));
    out[gid] += s * y[gid];
}
__global__ __launch_bounds__(256)
void relusq_bf16_kernel(const float* __restrict__ a, bf16_t* __restrict__ out,
                        long long n) {
    long long gid = (long long)blockIdx.x * blockDim.x + threadIdx.x;
    if (gid >= n) return;
    float t = a[gid];
    t = t > 0.0f ? t : 0.0f;
    out[gid] = (bf16_t)(t * t);
}
__global__ __launch_bounds__(256)
void gelu_bf16_kernel(const float* __restrict__ a, bf16_t* __restrict__ out,
                      long long n) {
    long long gid = (long long)blockIdx.x * blockDim.x + threadIdx.x;
    if (gid >= n) return;
    float t = a[gid];
    out[gid] = (bf16_t)(0.5f * t * (1.0f + erff(t * 0.70710678118654752f)));
}

// ---------------- causal softmax: fp32 in -> bf16 out ----------------
__global__ __launch_bounds__(256)
void softmax_kernel(const float* __restrict__ att, bf16_t* __restrict__ out,
                    int T, float scale) {
    long long row = blockIdx.x;            // (b*H + h)*T + i
    int i = (int)(row % T);
    int L = i + 1;
    const float* p = att + row * T;
    bf16_t* q = out + row * T;
    float mx = -3.0e38f;
    for (int j = threadIdx.x; j < L; j += blockDim.x) mx = fmaxf(mx, p[j] * scale);
    float m = blockMax(mx);
    float s = 0.0f;
    for (int j = threadIdx.x; j < L; j += blockDim.x) s += expf(p[j] * scale - m);
    float inv = 1.0f / blockSum(s);
    for (int j = threadIdx.x; j < T; j += blockDim.x)
        q[j] = (bf16_t)((j < L) ? expf(p[j] * scale - m) * inv : 0.0f);
}

// ---------------- WMMA bf16 GEMM, async double-buffered LDS ----------------
// C[M,N] = A[M,K] . W^T (+bias). A,W bf16 (W stored N-major, (N,K)).
// Block tile TM x TNv x TK, 8 waves; each wave computes 32 x (TNv/2).
#define TM  128
#define TK  32
#define TKP 40   // row stride 80B: 16B multiple (async B128) and bank-conflict free

template <int TNv>
__global__ __launch_bounds__(256)
void gemm_bf16_wmma(const bf16_t* __restrict__ A, const bf16_t* __restrict__ W,
                    const float* __restrict__ bias,
                    float* __restrict__ C32, bf16_t* __restrict__ C16,
                    int K, long long lda, long long ldb, long long ldc,
                    int accumulate, int batchH,
                    long long sAb, long long sAh,
                    long long sBb, long long sBh,
                    long long sCb, long long sCh) {
    constexpr int NB = TNv / 32;                       // B frags per wave (2 or 4)
    __shared__ __align__(16) bf16_t As[2][TM][TKP];
    __shared__ __align__(16) bf16_t Bs[2][TNv][TKP];

    const int z  = blockIdx.z;
    const int bz = z / batchH;
    const int hz = z % batchH;
    A += bz * sAb + hz * sAh;
    W += bz * sBb + hz * sBh;
    const long long coff = bz * sCb + hz * sCh;

    const int bm = blockIdx.y * TM;
    const int bn = blockIdx.x * TNv;

    const int tid  = threadIdx.x;
    const int lane = tid & 31;
    const int wave = tid >> 5;
    const int wm   = (wave >> 1) * 32;                 // 0,32,64,96
    const int wn   = (wave & 1) * (TNv / 2);           // 0, TNv/2

    const int half = lane >> 4;
    const int l15  = lane & 15;
    const int kbA  = half * 8;                         // A frag K base (ISA layout)
    const int kbB  = half * 8;                         // B frag K base (ISA layout)

    const int rowL = tid >> 2;                         // 0..63
    const int col8 = (tid & 3) * 8;                    // 16B chunks

    v8f acc[2][NB] = {};

    auto issueTile = [&](int k0, int buf) {
        #pragma unroll
        for (int r = 0; r < TM; r += 64)
            async_b128((unsigned)(size_t)&As[buf][rowL + r][col8],
                       A + (long long)(bm + rowL + r) * lda + (k0 + col8));
        #pragma unroll
        for (int r = 0; r < TNv; r += 64)
            async_b128((unsigned)(size_t)&Bs[buf][rowL + r][col8],
                       W + (long long)(bn + rowL + r) * ldb + (k0 + col8));
    };

    issueTile(0, 0);
    int buf = 0;
    for (int k0 = 0; k0 < K; k0 += TK, buf ^= 1) {
        async_wait0();                 // this wave's DMA for tile(buf) complete
        __syncthreads();               // all waves' tiles visible
        if (k0 + TK < K) issueTile(k0 + TK, buf ^ 1);  // prefetch next tile

        v16bf a[2], bf[NB];
        #pragma unroll
        for (int v = 0; v < 8; ++v) {
            int ka = (v < 4) ? (kbA + 2 * v) : (kbA + 16 + 2 * (v - 4));
            #pragma unroll
            for (int i = 0; i < 2; ++i) {
                a[i][2 * v]     = As[buf][wm + 16 * i + l15][ka];
                a[i][2 * v + 1] = As[buf][wm + 16 * i + l15][ka + 1];
            }
            int kb = (v < 4) ? (kbB + 2 * v) : (kbB + 16 + 2 * (v - 4));
            #pragma unroll
            for (int j = 0; j < NB; ++j) {
                bf[j][2 * v]     = Bs[buf][wn + 16 * j + l15][kb];
                bf[j][2 * v + 1] = Bs[buf][wn + 16 * j + l15][kb + 1];
            }
        }
        #pragma unroll
        for (int i = 0; i < 2; ++i)
            #pragma unroll
            for (int j = 0; j < NB; ++j)
                acc[i][j] = __builtin_amdgcn_wmma_f32_16x16x32_bf16(
                    false, a[i], false, bf[j], (short)0, acc[i][j], false, false);
        __syncthreads();               // readers done before buf is refilled
    }

    #pragma unroll
    for (int i = 0; i < 2; ++i)
        #pragma unroll
        for (int j = 0; j < NB; ++j)
            #pragma unroll
            for (int r = 0; r < 8; ++r) {
                int m = bm + wm + 16 * i + r + (half ? 8 : 0);
                int n = bn + wn + 16 * j + l15;
                float vv = acc[i][j][r];
                if (bias) vv += bias[n];
                long long o = coff + (long long)m * ldc + n;
                if (C32) { if (accumulate) C32[o] += vv; else C32[o] = vv; }
                else     { C16[o] = (bf16_t)vv; }
            }
}

// ---------------- host-side helpers ----------------
static inline void launch_gemm(hipStream_t s,
                               const bf16_t* A, const bf16_t* W, const float* bias,
                               float* C32, bf16_t* C16,
                               int M, int N, int K,
                               long long lda, long long ldb, long long ldc,
                               int acc,
                               int batches = 1, int batchH = 1,
                               long long sAb = 0, long long sAh = 0,
                               long long sBb = 0, long long sBh = 0,
                               long long sCb = 0, long long sCh = 0) {
    if (N % 128 == 0) {
        dim3 grid(N / 128, M / TM, batches);
        gemm_bf16_wmma<128><<<grid, 256, 0, s>>>(A, W, bias, C32, C16, K,
                                                 lda, ldb, ldc, acc, batchH,
                                                 sAb, sAh, sBb, sBh, sCb, sCh);
    } else {
        dim3 grid(N / 64, M / TM, batches);
        gemm_bf16_wmma<64><<<grid, 256, 0, s>>>(A, W, bias, C32, C16, K,
                                                lda, ldb, ldc, acc, batchH,
                                                sAb, sAh, sBb, sBh, sCb, sCh);
    }
}

extern "C" void kernel_launch(void* const* d_in, const int* in_sizes, int n_in,
                              void* d_out, int out_size, void* d_ws, size_t ws_size,
                              hipStream_t stream) {
    (void)in_sizes; (void)n_in; (void)out_size; (void)ws_size;
    const int B = BBATCH, T = TSEQ, D = DMODEL, H = NHEAD, HD = HEADD,
              FF = FFDIM, V = VOCAB;
    const long long BT  = (long long)B * T;        // 2048
    const long long BTD = BT * D;                  // 1.57M
    const long long BHTT = (long long)B * H * T * T;

    auto F = [&](int i) { return (const float*)d_in[i]; };
    const int* idx = (const int*)d_in[0];

    // byte bump allocator over d_ws (same sequence every call -> deterministic)
    char* base = (char*)d_ws;
    size_t off = 0;
    auto alloc = [&](size_t bytes) -> void* {
        off = (off + 255) & ~(size_t)255;
        void* p = base + off;
        off += bytes;
        return p;
    };

    const int EB = 256;
    auto gridOf = [&](long long n) { return (unsigned)((n + EB - 1) / EB); };

    // ---- convert all GEMM weights to bf16 once per launch ----
    auto conv = [&](const float* src, long long n) -> bf16_t* {
        bf16_t* dst = (bf16_t*)alloc((size_t)n * 2);
        f32_to_bf16_kernel<<<gridOf(n), EB, 0, stream>>>(src, dst, n);
        return dst;
    };
    bf16_t *wWk[4], *wWv[4], *wWr[4], *wWo[4], *wfWk[4], *wfWr[4], *wfWv[4];
    for (int L = 0; L < 4; ++L) {
        int p = 3 + L * 18;
        wWk[L]  = conv(F(p + 9),  (long long)D * D);
        wWv[L]  = conv(F(p + 10), (long long)D * D);
        wWr[L]  = conv(F(p + 11), (long long)D * D);
        wWo[L]  = conv(F(p + 12), (long long)D * D);
        wfWk[L] = conv(F(p + 15), (long long)FF * D);
        wfWr[L] = conv(F(p + 16), (long long)D * D);
        wfWv[L] = conv(F(p + 17), (long long)D * FF);
    }
    bf16_t *wqkv[2], *wout[2], *wff1[2], *wff2[2];
    for (int L = 0; L < 2; ++L) {
        int p = 75 + L * 12;
        wqkv[L] = conv(F(p + 4),  (long long)3 * D * D);
        wout[L] = conv(F(p + 6),  (long long)D * D);
        wff1[L] = conv(F(p + 8),  (long long)FF * D);
        wff2[L] = conv(F(p + 10), (long long)D * FF);
    }
    bf16_t* whead = conv(F(101), (long long)V * D);

    // ---- activation buffers ----
    float*  x     = (float*) alloc((size_t)BTD * 4);   // residual stream fp32
    bf16_t* h     = (bf16_t*)alloc((size_t)BTD * 2);   // LN out
    bf16_t* xk    = (bf16_t*)alloc((size_t)BTD * 2);
    bf16_t* xv    = (bf16_t*)alloc((size_t)BTD * 2);
    bf16_t* xr    = (bf16_t*)alloc((size_t)BTD * 2);
    float*  kb    = (float*) alloc((size_t)BTD * 4);
    float*  vb    = (float*) alloc((size_t)BTD * 4);
    float*  rb    = (float*) alloc((size_t)BTD * 4);
    float*  yb    = (float*) alloc((size_t)BTD * 4);
    bf16_t* rybf  = (bf16_t*)alloc((size_t)BTD * 2);   // sigmoid(r)*y / attn-out
    float*  kvf   = (float*) alloc((size_t)BTD * 4);
    float*  ffb   = (float*) alloc((size_t)BT * FF * 4);
    bf16_t* ffbf  = (bf16_t*)alloc((size_t)BT * FF * 2);
    bf16_t* qkvb  = (bf16_t*)alloc((size_t)BT * 3 * D * 2);
    bf16_t* vtb   = (bf16_t*)alloc((size_t)BTD * 2);   // V transposed (B*H, HD, T)
    float*  attf  = (float*) alloc((size_t)BHTT * 4);
    bf16_t* attbf = (bf16_t*)alloc((size_t)BHTT * 2);

    // ---- embedding + pos ----
    embed_kernel<<<gridOf(BTD), EB, 0, stream>>>(idx, F(1), F(2), x, T, D, BTD);

    // ---- RWKV layers ----
    for (int L = 0; L < 4; ++L) {
        int p = 3 + L * 18;
        // time mixing
        ln_kernel<<<(unsigned)BT, EB, 0, stream>>>(x, F(p + 0), F(p + 1), h, D);
        mix_kernel<<<gridOf(BTD), EB, 0, stream>>>(h, F(p + 6), F(p + 7), F(p + 8),
                                                   xk, xv, xr, T, D, BTD);
        launch_gemm(stream, xk, wWk[L], nullptr, kb, nullptr, (int)BT, D, D, D, D, D, 0);
        launch_gemm(stream, xv, wWv[L], nullptr, vb, nullptr, (int)BT, D, D, D, D, D, 0);
        launch_gemm(stream, xr, wWr[L], nullptr, rb, nullptr, (int)BT, D, D, D, D, D, 0);
        wkv_kernel<<<gridOf((long long)B * D), EB, 0, stream>>>(kb, vb, F(p + 4), F(p + 5),
                                                                yb, B, T, D);
        sigmul_bf16_kernel<<<gridOf(BTD), EB, 0, stream>>>(rb, yb, rybf, BTD);
        launch_gemm(stream, rybf, wWo[L], nullptr, x, nullptr, (int)BT, D, D, D, D, D, 1);
        // channel mixing
        ln_kernel<<<(unsigned)BT, EB, 0, stream>>>(x, F(p + 2), F(p + 3), h, D);
        mix_kernel<<<gridOf(BTD), EB, 0, stream>>>(h, F(p + 13), F(p + 14), nullptr,
                                                   xk, xr, nullptr, T, D, BTD);
        launch_gemm(stream, xk, wfWk[L], nullptr, ffb, nullptr, (int)BT, FF, D, D, D, FF, 0);
        relusq_bf16_kernel<<<gridOf(BT * FF), EB, 0, stream>>>(ffb, ffbf, BT * FF);
        launch_gemm(stream, ffbf, wfWv[L], nullptr, kvf, nullptr, (int)BT, D, FF, FF, FF, D, 0);
        launch_gemm(stream, xr, wfWr[L], nullptr, rb, nullptr, (int)BT, D, D, D, D, D, 0);
        sigmul_acc_kernel<<<gridOf(BTD), EB, 0, stream>>>(rb, kvf, x, BTD);
    }

    // ---- attention layers ----
    for (int L = 0; L < 2; ++L) {
        int p = 75 + L * 12;
        ln_kernel<<<(unsigned)BT, EB, 0, stream>>>(x, F(p + 0), F(p + 1), h, D);
        launch_gemm(stream, h, wqkv[L], F(p + 5), nullptr, qkvb,
                    (int)BT, 3 * D, D, D, D, 3 * D, 0);
        vtrans_kernel<<<gridOf(BTD), EB, 0, stream>>>(qkvb, vtb, T, D, H, HD, BTD);
        // att = Q K^T (24 batched GEMMs: M=T, N=T, K=HD)
        launch_gemm(stream, qkvb /*Q*/, qkvb + D /*K*/, nullptr, attf, nullptr,
                    T, T, HD, 3LL * D, 3LL * D, T, 0,
                    B * H, H,
                    (long long)T * 3 * D, HD,
                    (long long)T * 3 * D, HD,
                    (long long)H * T * T, (long long)T * T);
        softmax_kernel<<<(unsigned)((long long)B * H * T), EB, 0, stream>>>(
            attf, attbf, T, 0.125f /*1/sqrt(64)*/);
        // out = att V (M=T, N=HD, K=T; V^T pre-transposed, N-major)
        launch_gemm(stream, attbf, vtb, nullptr, nullptr, rybf,
                    T, HD, T, T, T, D, 0,
                    B * H, H,
                    (long long)H * T * T, (long long)T * T,
                    (long long)H * HD * T, (long long)HD * T,
                    (long long)T * D, HD);
        launch_gemm(stream, rybf, wout[L], F(p + 7), x, nullptr,
                    (int)BT, D, D, D, D, D, 1);
        // FFN
        ln_kernel<<<(unsigned)BT, EB, 0, stream>>>(x, F(p + 2), F(p + 3), h, D);
        launch_gemm(stream, h, wff1[L], F(p + 9), ffb, nullptr,
                    (int)BT, FF, D, D, D, FF, 0);
        gelu_bf16_kernel<<<gridOf(BT * FF), EB, 0, stream>>>(ffb, ffbf, BT * FF);
        launch_gemm(stream, ffbf, wff2[L], F(p + 11), x, nullptr,
                    (int)BT, D, FF, FF, FF, D, 1);
    }

    // ---- final LN + head ----
    ln_kernel<<<(unsigned)BT, EB, 0, stream>>>(x, F(99), F(100), h, D);
    launch_gemm(stream, h, whead, F(102), (float*)d_out, nullptr,
                (int)BT, V, D, D, D, V, 0);
}
